// Regressor_83288005804790
// MI455X (gfx1250) — matrix operand order repaired and checked by reading, not compiled, not run on hardware
//
#include <hip/hip_runtime.h>

#define BB 256
#define TT 256

typedef __bf16 v16bf __attribute__((ext_vector_type(16)));
typedef float  v8f   __attribute__((ext_vector_type(8)));

union Frag {
    uint4        q[2];
    unsigned int u[8];
    v16bf        v;
};
union Acc {
    v8f   v;
    float f[8];
};

__device__ __forceinline__ unsigned short f2bf(float f) {
    unsigned int x = __builtin_bit_cast(unsigned int, f);
    x += 0x7fffu + ((x >> 16) & 1u);          // round-to-nearest-even
    return (unsigned short)(x >> 16);
}
__device__ __forceinline__ float bf2f(unsigned short h) {
    unsigned int u = ((unsigned int)h) << 16;
    return __builtin_bit_cast(float, u);
}
// branch-free fast transcendentals using raw v_exp_f32 / v_rcp_f32
// (saturate correctly at +/-inf; ~1 ulp rcp is fine vs bf16 matmul inputs)
__device__ __forceinline__ float sigm(float x) {
    return __builtin_amdgcn_rcpf(1.0f + __expf(-x));
}
__device__ __forceinline__ float tanh_fast(float x) {
    float e = __expf(2.0f * x);
    return 1.0f - 2.0f * __builtin_amdgcn_rcpf(e + 1.0f);
}

// ---------------------------------------------------------------------------
// NSPT extractor: 3 strided convs + relu, mean over length -> feat[B][16]
// ---------------------------------------------------------------------------
__global__ __launch_bounds__(64) void nspt_feat_kernel(
    const float* __restrict__ nspt,
    const float* __restrict__ w1, const float* __restrict__ b1,
    const float* __restrict__ w2, const float* __restrict__ b2,
    const float* __restrict__ w3, const float* __restrict__ b3,
    float* __restrict__ feat)
{
    __shared__ float xb[3 * 128];
    __shared__ float y1[8 * 64];
    __shared__ float y2[8 * 32];
    __shared__ float y3[16 * 16];
    const int b = blockIdx.x, tid = threadIdx.x;
    for (int i = tid; i < 3 * 128; i += 64) xb[i] = nspt[(size_t)b * 384 + i];
    __syncthreads();
    for (int i = tid; i < 8 * 64; i += 64) {
        int o = i >> 6, l = i & 63;
        float s = b1[o];
        for (int ic = 0; ic < 3; ++ic)
            #pragma unroll
            for (int kk = 0; kk < 3; ++kk) {
                int p = 2 * l - 1 + kk;
                if (p >= 0 && p < 128) s += w1[(o * 3 + ic) * 3 + kk] * xb[ic * 128 + p];
            }
        y1[i] = fmaxf(s, 0.f);
    }
    __syncthreads();
    for (int i = tid; i < 8 * 32; i += 64) {
        int o = i >> 5, l = i & 31;
        float s = b2[o];
        for (int ic = 0; ic < 8; ++ic)
            #pragma unroll
            for (int kk = 0; kk < 3; ++kk) {
                int p = 2 * l - 1 + kk;
                if (p >= 0 && p < 64) s += w2[(o * 8 + ic) * 3 + kk] * y1[ic * 64 + p];
            }
        y2[i] = fmaxf(s, 0.f);
    }
    __syncthreads();
    for (int i = tid; i < 16 * 16; i += 64) {
        int o = i >> 4, l = i & 15;
        float s = b3[o];
        for (int ic = 0; ic < 8; ++ic)
            #pragma unroll
            for (int kk = 0; kk < 3; ++kk) {
                int p = 2 * l - 1 + kk;
                if (p >= 0 && p < 32) s += w3[(o * 8 + ic) * 3 + kk] * y2[ic * 32 + p];
            }
        y3[i] = fmaxf(s, 0.f);
    }
    __syncthreads();
    if (tid < 16) {
        float s = 0.f;
        #pragma unroll
        for (int l = 0; l < 16; ++l) s += y3[tid * 16 + l];
        feat[(size_t)b * 16 + tid] = s * (1.0f / 16.0f);
    }
}

// ---------------------------------------------------------------------------
// Build per-step input x, K-padded to 32, time-major bf16: xpad[T][B][32]
// ---------------------------------------------------------------------------
__global__ __launch_bounds__(256) void xpad_kernel(
    const float* __restrict__ inp, const float* __restrict__ Wds,
    const float* __restrict__ bds, const float* __restrict__ feat,
    unsigned short* __restrict__ xpad)
{
    int idx = blockIdx.x * 256 + threadIdx.x;   // b*T + t
    int b = idx >> 8;
    int t = idx & 255;
    const float* xi = inp + ((size_t)b * TT + t) * 8;
    float v[32];
    #pragma unroll
    for (int d = 0; d < 4; ++d) {
        float s = bds[d];
        #pragma unroll
        for (int f = 0; f < 7; ++f) s += Wds[d * 7 + f] * xi[f];
        v[d] = s;
    }
    v[4] = xi[7];
    #pragma unroll
    for (int j = 0; j < 16; ++j) v[5 + j] = feat[b * 16 + j];
    #pragma unroll
    for (int j = 21; j < 32; ++j) v[j] = 0.f;
    unsigned short* dst = xpad + ((size_t)t * BB + b) * 32;
    #pragma unroll
    for (int j = 0; j < 32; ++j) dst[j] = f2bf(v[j]);
}

// ---------------------------------------------------------------------------
// Pre-swizzle cell0 weights into WMMA B-fragment layout.
// Combined [K=288 x N=512]: n<256 -> W_i2h0, n>=256 -> W_i2o0.
// K map: 0..20 = x cols, 21..31 = zero pad, 32..287 = h cols 21..276.
// Fragment f = nt*9+kc, element e = ln*16+idx:
//   k = kc*32 + (ln/16)*16 + idx, n = nt*16 + ln%16
// ---------------------------------------------------------------------------
__global__ __launch_bounds__(256) void frag0_kernel(
    const float* __restrict__ Wh, const float* __restrict__ Wo,
    unsigned short* __restrict__ frag)
{
    int f = blockIdx.x;              // 0..287 = nt*9 + kc
    int nt = f / 9, kc = f - nt * 9;
    int e0 = threadIdx.x * 2;
    #pragma unroll
    for (int q = 0; q < 2; ++q) {
        int e = e0 + q;
        int ln = e >> 4, idx = e & 15;
        int k = kc * 32 + ((ln >> 4) << 4) + idx;
        int n = nt * 16 + (ln & 15);
        float w = 0.f;
        int col = (k < 21) ? k : ((k >= 32) ? (k - 11) : -1);
        if (col >= 0) w = (n < 256) ? Wh[n * 277 + col] : Wo[(n - 256) * 277 + col];
        frag[(size_t)f * 512 + e] = f2bf(w);
    }
}

// cell1 weights [K=512 x N=256], same fragment layout (f = nt*16 + kc)
__global__ __launch_bounds__(256) void frag1_kernel(
    const float* __restrict__ W1, unsigned short* __restrict__ frag)
{
    int f = blockIdx.x;
    int nt = f >> 4, kc = f & 15;
    int e0 = threadIdx.x * 2;
    #pragma unroll
    for (int q = 0; q < 2; ++q) {
        int e = e0 + q;
        int ln = e >> 4, idx = e & 15;
        int k = kc * 32 + ((ln >> 4) << 4) + idx;
        int n = nt * 16 + (ln & 15);
        frag[(size_t)f * 512 + e] = f2bf(W1[n * 512 + k]);
    }
}

// ---------------------------------------------------------------------------
// Opaque (non-hoistable) fragment loaders: one base address + immediate
// offsets for all tiles of this k-chunk; prefetch next k-chunk; single wait.
// cell0 strides: +9216 B next a-tile, +147456 B to g-tile block.
// ---------------------------------------------------------------------------
__device__ __forceinline__ void load_frags_c0(
    Frag& bA0, Frag& bA1, Frag& bG0, Frag& bG1, const unsigned short* base)
{
    asm volatile(
        "global_load_b128 %0, %8, off\n\t"
        "global_load_b128 %1, %8, off offset:16\n\t"
        "global_load_b128 %2, %8, off offset:9216\n\t"
        "global_load_b128 %3, %8, off offset:9232\n\t"
        "global_load_b128 %4, %8, off offset:147456\n\t"
        "global_load_b128 %5, %8, off offset:147472\n\t"
        "global_load_b128 %6, %8, off offset:156672\n\t"
        "global_load_b128 %7, %8, off offset:156688\n\t"
        "global_prefetch_b8 %8, off offset:1024\n\t"
        "s_wait_loadcnt 0x0"
        : "=&v"(bA0.q[0]), "=&v"(bA0.q[1]), "=&v"(bA1.q[0]), "=&v"(bA1.q[1]),
          "=&v"(bG0.q[0]), "=&v"(bG0.q[1]), "=&v"(bG1.q[0]), "=&v"(bG1.q[1])
        : "v"(base)
        : "memory");
}
// cell1 stride: +16384 B next tile
__device__ __forceinline__ void load_frags_c1(
    Frag& b0, Frag& b1, const unsigned short* base)
{
    asm volatile(
        "global_load_b128 %0, %4, off\n\t"
        "global_load_b128 %1, %4, off offset:16\n\t"
        "global_load_b128 %2, %4, off offset:16384\n\t"
        "global_load_b128 %3, %4, off offset:16400\n\t"
        "global_prefetch_b8 %4, off offset:1024\n\t"
        "s_wait_loadcnt 0x0"
        : "=&v"(b0.q[0]), "=&v"(b0.q[1]), "=&v"(b1.q[0]), "=&v"(b1.q[1])
        : "v"(base)
        : "memory");
}

// ---------------------------------------------------------------------------
// Fused 2-cell recurrence. 16 blocks x 256 threads (8 wave32).
// c-states in registers (native C/D layout), h-states in LDS as bf16 forming
// next step's WMMA A matrix. Wave w owns features [32w, 32w+32).
// ---------------------------------------------------------------------------
__global__ __launch_bounds__(256) void lstm_fused_kernel(
    const unsigned short* __restrict__ xpad,
    const unsigned short* __restrict__ fragW0,
    const unsigned short* __restrict__ fragW1,
    const float* __restrict__ bh0, const float* __restrict__ bo0,
    const float* __restrict__ bh1,
    const float* __restrict__ Wo1, const float* __restrict__ bo1,
    float* __restrict__ out)
{
    __shared__ __align__(16) unsigned short comb0[16 * 288]; // [m][k] bf16
    __shared__ __align__(16) unsigned short comb1[16 * 512]; // [m][k] bf16
    __shared__ float g1p[16][16];
    __shared__ float g1v[16];
    __shared__ float wo1s[512];

    const int tid  = threadIdx.x;
    const int lane = tid & 31;
    const int wave = tid >> 5;          // 0..7
    const int b0   = blockIdx.x * 16;
    const int nloc = lane & 15;         // N col / A row within tile
    const int hh   = lane >> 4;         // K-half select
    const int mhi  = hh << 3;           // C/D: lanes 16-31 hold M+8

    for (int i = tid; i < 16 * 288; i += 256) comb0[i] = 0;
    for (int i = tid; i < 16 * 512; i += 256) comb1[i] = 0;
    for (int i = tid; i < 512; i += 256) wo1s[i] = Wo1[i];

    float c0r[2][8], c1r[2][8];
    #pragma unroll
    for (int p = 0; p < 2; ++p)
        #pragma unroll
        for (int r = 0; r < 8; ++r) { c0r[p][r] = 0.f; c1r[p][r] = 0.f; }

    float biasA0[2], biasG0[2], biasA1[2];
    #pragma unroll
    for (int p = 0; p < 2; ++p) {
        int n = (2 * wave + p) * 16 + nloc;
        biasA0[p] = bh0[n];
        biasG0[p] = bo0[n];
        biasA1[p] = bh1[n];
    }
    const float bO1 = bo1[0];

    // per-wave fragment base pointers (advance by 512 shorts per k-chunk)
    const unsigned short* w0base0 = fragW0 + (size_t)(2 * wave) * 9 * 512 + lane * 16;
    const unsigned short* w1base0 = fragW1 + (size_t)(2 * wave) * 16 * 512 + lane * 16;

    __syncthreads();

    for (int t = 0; t < TT; ++t) {
        // stage x_t into comb0 K cols 0..31 (one u32 per thread)
        {
            int m = tid >> 4, cp = tid & 15;
            const unsigned int* src =
                (const unsigned int*)(xpad + ((size_t)t * BB + (b0 + m)) * 32);
            *(unsigned int*)(&comb0[m * 288 + cp * 2]) = src[cp];
        }
        __syncthreads();

        // ---- cell 0: [16x288] x [288x512] (a | g), bias folded into C ----
        Acc accA[2], accG[2];
        #pragma unroll
        for (int p = 0; p < 2; ++p)
            #pragma unroll
            for (int r = 0; r < 8; ++r) { accA[p].f[r] = biasA0[p]; accG[p].f[r] = biasG0[p]; }

        const unsigned short* w0p = w0base0;
        for (int kc = 0; kc < 9; ++kc, w0p += 512) {
            Frag a;   // A 16x32 bf16: row = nloc, VGPR j -> K = 16*hh + 2j, 2j+1 (contiguous)
            #pragma unroll
            for (int j = 0; j < 8; ++j) {
                int k0 = hh * 16 + (j << 1);
                a.u[j] = *(const unsigned int*)(&comb0[nloc * 288 + kc * 32 + k0]);
            }
            Frag bA0, bA1, bG0, bG1;
            load_frags_c0(bA0, bA1, bG0, bG1, w0p);
            accA[0].v = __builtin_amdgcn_wmma_f32_16x16x32_bf16(
                false, a.v, false, bA0.v, (short)0, accA[0].v, false, false);
            accA[1].v = __builtin_amdgcn_wmma_f32_16x16x32_bf16(
                false, a.v, false, bA1.v, (short)0, accA[1].v, false, false);
            accG[0].v = __builtin_amdgcn_wmma_f32_16x16x32_bf16(
                false, a.v, false, bG0.v, (short)0, accG[0].v, false, false);
            accG[1].v = __builtin_amdgcn_wmma_f32_16x16x32_bf16(
                false, a.v, false, bG1.v, (short)0, accG[1].v, false, false);
        }
        __syncthreads();   // all comb0 reads complete

        // elementwise cell0: c0' = ig*tanh(c0)+(1-ig)*tanh(a); h0' = ig*tanh(c0')
        #pragma unroll
        for (int p = 0; p < 2; ++p) {
            int n = (2 * wave + p) * 16 + nloc;
            #pragma unroll
            for (int r = 0; r < 8; ++r) {
                int m = mhi + r;
                float av = accA[p].f[r];
                float gv = accG[p].f[r];
                float ig = sigm(gv);
                float cn = ig * tanh_fast(c0r[p][r]) + (1.0f - ig) * tanh_fast(av);
                c0r[p][r] = cn;
                comb0[m * 288 + 32 + n]  = f2bf(ig * tanh_fast(cn)); // h0 -> next A
                comb1[m * 512 + n]       = f2bf(gv);                 // o0 = raw g
            }
        }
        __syncthreads();   // comb1 o0-part + comb0 h-part visible

        // g1 partials (deterministic two-stage reduction)
        {
            int m = tid & 15, seg = tid >> 4;
            const unsigned short* row = &comb1[m * 512 + seg * 32];
            const float* wr = &wo1s[seg * 32];
            float s = 0.f;
            #pragma unroll
            for (int k = 0; k < 32; ++k) s += bf2f(row[k]) * wr[k];
            g1p[m][seg] = s;
        }

        // ---- cell 1: [16x512] x [512x256] ----
        Acc acc1[2];
        #pragma unroll
        for (int p = 0; p < 2; ++p)
            #pragma unroll
            for (int r = 0; r < 8; ++r) acc1[p].f[r] = biasA1[p];

        const unsigned short* w1p = w1base0;
        for (int kc = 0; kc < 16; ++kc, w1p += 512) {
            Frag a;
            #pragma unroll
            for (int j = 0; j < 8; ++j) {
                int k0 = hh * 16 + (j << 1);
                a.u[j] = *(const unsigned int*)(&comb1[nloc * 512 + kc * 32 + k0]);
            }
            Frag b0f, b1f;
            load_frags_c1(b0f, b1f, w1p);
            acc1[0].v = __builtin_amdgcn_wmma_f32_16x16x32_bf16(
                false, a.v, false, b0f.v, (short)0, acc1[0].v, false, false);
            acc1[1].v = __builtin_amdgcn_wmma_f32_16x16x32_bf16(
                false, a.v, false, b1f.v, (short)0, acc1[1].v, false, false);
        }
        __syncthreads();   // comb1 reads + g1p writes complete

        if (tid < 16) {
            float s = bO1;
            #pragma unroll
            for (int seg = 0; seg < 16; ++seg) s += g1p[tid][seg];
            g1v[tid] = s;
            out[(size_t)(b0 + tid) * TT + t] = s;   // per-step output o1 = g1
        }
        __syncthreads();   // g1v ready

        // elementwise cell1; h1' -> comb1 K cols 256..511 for next step
        #pragma unroll
        for (int p = 0; p < 2; ++p) {
            int n = (2 * wave + p) * 16 + nloc;
            #pragma unroll
            for (int r = 0; r < 8; ++r) {
                int m = mhi + r;
                float ig = sigm(g1v[m]);
                float cn = ig * tanh_fast(c1r[p][r]) + (1.0f - ig) * tanh_fast(acc1[p].f[r]);
                c1r[p][r] = cn;
                comb1[m * 512 + 256 + n] = f2bf(ig * tanh_fast(cn));
            }
        }
        __syncthreads();   // h1 writes done before next step's reads
    }
}

// ---------------------------------------------------------------------------
extern "C" void kernel_launch(void* const* d_in, const int* in_sizes, int n_in,
                              void* d_out, int out_size, void* d_ws, size_t ws_size,
                              hipStream_t stream) {
    (void)in_sizes; (void)n_in; (void)out_size; (void)ws_size;
    const float* inp  = (const float*)d_in[0];
    const float* nspt = (const float*)d_in[1];
    const float* Wds  = (const float*)d_in[2];
    const float* bds  = (const float*)d_in[3];
    const float* c1w  = (const float*)d_in[4];
    const float* c1b  = (const float*)d_in[5];
    const float* c2w  = (const float*)d_in[6];
    const float* c2b  = (const float*)d_in[7];
    const float* c3w  = (const float*)d_in[8];
    const float* c3b  = (const float*)d_in[9];
    const float* Wh0  = (const float*)d_in[10];
    const float* bh0  = (const float*)d_in[11];
    const float* Wo0  = (const float*)d_in[12];
    const float* bo0  = (const float*)d_in[13];
    const float* Wh1  = (const float*)d_in[14];
    const float* bh1  = (const float*)d_in[15];
    const float* Wo1  = (const float*)d_in[16];
    const float* bo1  = (const float*)d_in[17];

    char* ws = (char*)d_ws;
    float*          feat  = (float*)ws;                                       // 16 KB
    unsigned short* xpad  = (unsigned short*)(ws + 16384);                    // 4 MB
    unsigned short* frag0 = (unsigned short*)(ws + 16384 + 4194304);          // 288 KB
    unsigned short* frag1 = (unsigned short*)(ws + 16384 + 4194304 + 294912); // 256 KB

    nspt_feat_kernel<<<BB, 64, 0, stream>>>(nspt, c1w, c1b, c2w, c2b, c3w, c3b, feat);
    xpad_kernel<<<(BB * TT) / 256, 256, 0, stream>>>(inp, Wds, bds, feat, xpad);
    frag0_kernel<<<288, 256, 0, stream>>>(Wh0, Wo0, frag0);
    frag1_kernel<<<256, 256, 0, stream>>>(Wh1, frag1);
    lstm_fused_kernel<<<16, 256, 0, stream>>>(xpad, frag0, frag1,
                                              bh0, bo0, bh1, Wo1, bo1,
                                              (float*)d_out);
}
